// AttHead_19361712570491
// MI455X (gfx1250) — compile-verified, matrix operand order checked by reading, not run on hardware
//
#include <hip/hip_runtime.h>
#include <hip/hip_bf16.h>
#include <math.h>

// ---------------- problem constants ----------------
#define B_   256
#define C_   512
#define HW_  256      // H*W = 8*32
#define A_   512
#define HID_ 512
#define NC_  97
#define T_   26
#define INP  640      // IN = C + NC = 609, padded to multiple of 32
#define G3   1536     // 3*HID
#define NCP  112      // NC padded to multiple of 16

typedef __bf16 bf16;
typedef __bf16 v16bf __attribute__((ext_vector_type(16)));
typedef __bf16 v8bf  __attribute__((ext_vector_type(8)));
typedef float  v8f   __attribute__((ext_vector_type(8)));
typedef unsigned int v4u __attribute__((ext_vector_type(4)));
typedef int  v8i_ __attribute__((ext_vector_type(8)));
typedef int  v4i_ __attribute__((ext_vector_type(4)));

// ---------------- fast transcendental ----------------
__device__ inline float fast_tanh(float x) {
#if __has_builtin(__builtin_amdgcn_tanhf)
    return __builtin_amdgcn_tanhf(x);      // V_TANH_F32 (TRANS)
#else
    return tanhf(x);
#endif
}

// ---------------- WMMA helpers ----------------
__device__ inline v8f wmma_bf16(v16bf a, v16bf b, v8f c) {
    return __builtin_amdgcn_wmma_f32_16x16x32_bf16(
        false, a, false, b, (short)0, c, false, false);
}

// k-contiguous bf16 fragment load (works for global or LDS base pointers).
// lanes 0-15: row (lane&15), K = k0..k0+7 / k0+16..k0+23
// lanes 16-31: row (lane&15), K = k0+8..k0+15 / k0+24..k0+31
__device__ inline v16bf load_frag(const bf16* __restrict__ base, int ld,
                                  int row, int k0, int lane) {
    const bf16* p = base + (size_t)row * ld + k0 + ((lane >> 4) << 3);
    v8bf lo = *(const v8bf*)p;
    v8bf hi = *(const v8bf*)(p + 16);
    v16bf r;
#pragma unroll
    for (int i = 0; i < 8; ++i) { r[i] = lo[i]; r[8 + i] = hi[i]; }
    return r;
}

// ---------------- TDM: 2D tile load -> LDS (6-arg builtin on this toolchain) ----------------
// Descriptor per CDNA5 ISA.  data_size = 8 bytes.
#if __has_builtin(__builtin_amdgcn_tensor_load_to_lds)
#define HAVE_TDM 1
__device__ inline void tdm_load_2d(unsigned lds_off, const void* gaddr,
                                   unsigned rows, unsigned row_qw, unsigned stride_qw) {
    unsigned long long ga = (unsigned long long)(size_t)gaddr;
    v4u g0;
    g0[0] = 1u;                                            // count=1 (valid), user mode
    g0[1] = lds_off;                                       // lds_addr
    g0[2] = (unsigned)(ga & 0xffffffffu);                  // global_addr[31:0]
    g0[3] = (unsigned)((ga >> 32) & 0x01ffffffu) | (2u << 30); // addr[56:32] | type=2
    v8i_ g1;
    g1[0] = (int)(3u << 16);                               // wg_mask=0, data_size=3 (8B)
    g1[1] = (int)((row_qw & 0xffffu) << 16);               // tensor_dim0 lo16
    g1[2] = (int)(((row_qw >> 16) & 0xffffu) | ((rows & 0xffffu) << 16)); // d0 hi | d1 lo
    g1[3] = (int)(((rows >> 16) & 0xffffu) | ((row_qw & 0xffffu) << 16)); // d1 hi | tile_dim0
    g1[4] = (int)(rows & 0xffffu);                         // tile_dim1 (tile_dim2=0)
    g1[5] = (int)stride_qw;                                // tensor_dim0_stride lo32
    g1[6] = 0;                                             // stride hi | dim1_stride lo
    g1[7] = 0;
    v4i_ gz4 = {0, 0, 0, 0};
    v8i_ gz8 = {0, 0, 0, 0, 0, 0, 0, 0};
    __builtin_amdgcn_tensor_load_to_lds(g0, g1, gz4, gz4, gz8, 0);
}
#else
#define HAVE_TDM 0
#endif

// ---------------- setup kernels ----------------
__global__ __launch_bounds__(256)
void k_convert_pad(const float* __restrict__ src, bf16* __restrict__ dst,
                   int srcR, int srcC, int dstR, int dstC) {
    int i = blockIdx.x * 256 + threadIdx.x;
    if (i >= dstR * dstC) return;
    int r = i / dstC, c = i - r * dstC;
    float v = (r < srcR && c < srcC) ? src[r * srcC + c] : 0.f;
    dst[i] = (bf16)v;
}

__global__ __launch_bounds__(256)
void k_transpose_feats(const float* __restrict__ feats, bf16* __restrict__ featsT) {
    int i = blockIdx.x * 256 + threadIdx.x;     // B*HW*C threads
    int c   = i & (C_ - 1);
    int rem = i >> 9;
    int hw  = rem & (HW_ - 1);
    int b   = rem >> 8;
    featsT[i] = (bf16)feats[((size_t)b * C_ + c) * HW_ + hw];
}

__global__ __launch_bounds__(256)
void k_zero_h(float* __restrict__ h, bf16* __restrict__ h_bf) {
    int i = blockIdx.x * 256 + threadIdx.x;     // B*HID
    h[i] = 0.f;
    h_bf[i] = (bf16)0.f;
}

// ---------------- one-time: feat_x = Wf * feats_b + bf ----------------
// Block = (b, a-tile): TDM stages the 16x512 bf16 strip of Wf into LDS once,
// 8 waves each produce two 16x16 hw-tiles reading A frags from LDS.
// Output stored transposed: fxT[b][hw][a] bf16.  Grid: (32, B), 256 thr.
__global__ __launch_bounds__(256)
void k_featx(const bf16* __restrict__ Wf_bf, const bf16* __restrict__ featsT,
             const float* __restrict__ bf_, bf16* __restrict__ fxT) {
    __shared__ __align__(16) bf16 As[16 * C_];            // 16 KB
    int b   = blockIdx.y;
    int a0  = blockIdx.x << 4;                            // 32 a-tiles
    int wid = threadIdx.x >> 5, lane = threadIdx.x & 31;
#if HAVE_TDM
    if (wid == 0) {
        tdm_load_2d((unsigned)(size_t)&As[0], Wf_bf + (size_t)a0 * C_,
                    /*rows=*/16, /*row_qw=*/C_ * 2 / 8, /*stride_qw=*/C_ * 2 / 8);
        __builtin_amdgcn_s_wait_tensorcnt(0);
    }
#else
    {
        const v8bf* src = (const v8bf*)(Wf_bf + (size_t)a0 * C_);
        v8bf* dst = (v8bf*)As;
        for (int i = threadIdx.x; i < 16 * C_ / 8; i += 256) dst[i] = src[i];
    }
#endif
    __syncthreads();

    int hw0 = wid << 5;                                   // each wave: hw tiles hw0, hw0+16
    const bf16* Bp = featsT + (size_t)b * HW_ * C_;
    int rB = hw0 + (lane & 15);
    v8f acc0 = {}, acc1 = {};
#pragma unroll 2
    for (int k = 0; k < C_; k += 32) {
        v16bf a  = load_frag(As, C_, lane & 15, k, lane); // ds_load_b128 x2
        v16bf b0 = load_frag(Bp, C_, rB,      k, lane);
        v16bf b1 = load_frag(Bp, C_, rB + 16, k, lane);
        acc0 = wmma_bf16(a, b0, acc0);
        acc1 = wmma_bf16(a, b1, acc1);
    }
    int hi = lane >> 4, n = lane & 15;
    v8bf o0, o1;
#pragma unroll
    for (int r = 0; r < 8; ++r) {
        float bias = bf_[a0 + 8 * hi + r];
        o0[r] = (bf16)(acc0[r] + bias);
        o1[r] = (bf16)(acc1[r] + bias);
    }
    bf16* outb = fxT + (size_t)b * HW_ * C_ + a0 + 8 * hi;
    *(v8bf*)(outb + (size_t)(hw0 + n) * C_)      = o0;
    *(v8bf*)(outb + (size_t)(hw0 + 16 + n) * C_) = o1;
}

// ---------------- per-step: hq = h @ Wh^T + bh   [B, A] f32 ----------------
// 4-way N register blocking: wave = 16(b) x 64(a).  128 waves -> 16 blocks.
__global__ __launch_bounds__(256)
void k_hq(const bf16* __restrict__ h_bf, const bf16* __restrict__ Wh_bf,
          const float* __restrict__ bh, float* __restrict__ hq) {
    int tile = blockIdx.x * 8 + (threadIdx.x >> 5);   // 16 b-tiles x 8 n-groups
    int lane = threadIdx.x & 31;
    int b0 = (tile & 15) << 4;
    int n0 = (tile >> 4) << 6;
    int rA = b0 + (lane & 15);
    int rB = n0 + (lane & 15);
    v8f acc[4] = {{}, {}, {}, {}};
#pragma unroll 2
    for (int k = 0; k < HID_; k += 32) {
        v16bf a = load_frag(h_bf, HID_, rA, k, lane);
#pragma unroll
        for (int j = 0; j < 4; ++j) {
            v16bf bb = load_frag(Wh_bf, HID_, rB + 16 * j, k, lane);
            acc[j] = wmma_bf16(a, bb, acc[j]);
        }
    }
    int hi = lane >> 4, n = lane & 15;
#pragma unroll
    for (int j = 0; j < 4; ++j) {
        float bias = bh[n0 + 16 * j + n];
#pragma unroll
        for (int r = 0; r < 8; ++r)
            hq[(size_t)(b0 + 8 * hi + r) * A_ + n0 + 16 * j + n] = acc[j][r] + bias;
    }
}

// ---------------- per-step: attention + GRU-input assembly ----------------
__global__ __launch_bounds__(256)
void k_attn(const bf16* __restrict__ featsT, const bf16* __restrict__ fxT,
            const float* __restrict__ hq, const float* __restrict__ ws,
            const int* __restrict__ texts, int t, bf16* __restrict__ x_bf) {
    __shared__ float sc[HW_];
    __shared__ float red[256];
    int b = blockIdx.x, tid = threadIdx.x;
    int wid = tid >> 5, lane = tid & 31;

    // per-lane invariant ws / hq chunk (16 contiguous 'a' values per lane)
    int a0 = lane << 4;
    float wsr[16], hqr[16];
    {
        const float* hb = hq + (size_t)b * A_;
#pragma unroll
        for (int i = 0; i < 16; ++i) { wsr[i] = ws[a0 + i]; hqr[i] = hb[a0 + i]; }
    }

    // phase 1: score[hw] = sum_a ws[a]*tanh(feat_x + hq); vectorized 16B loads
    const bf16* fx = fxT + (size_t)b * HW_ * C_;
    for (int hw = wid; hw < HW_; hw += 8) {
        const bf16* row = fx + (size_t)hw * A_ + a0;
        v8bf x0 = *(const v8bf*)row;
        v8bf x1 = *(const v8bf*)(row + 8);
        float p = 0.f;
#pragma unroll
        for (int i = 0; i < 8; ++i) p += wsr[i]     * fast_tanh((float)x0[i] + hqr[i]);
#pragma unroll
        for (int i = 0; i < 8; ++i) p += wsr[8 + i] * fast_tanh((float)x1[i] + hqr[8 + i]);
#pragma unroll
        for (int off = 16; off; off >>= 1) p += __shfl_down(p, off, 32);
        if (lane == 0) sc[hw] = p;
    }
    __syncthreads();

    // phase 2: softmax over 256 spatial positions
    float v = sc[tid];
    red[tid] = v; __syncthreads();
    for (int s = 128; s; s >>= 1) { if (tid < s) red[tid] = fmaxf(red[tid], red[tid + s]); __syncthreads(); }
    float mx = red[0]; __syncthreads();
    float e = __expf(v - mx);
    red[tid] = e; __syncthreads();
    for (int s = 128; s; s >>= 1) { if (tid < s) red[tid] += red[tid + s]; __syncthreads(); }
    float inv = 1.f / red[0];
    __syncthreads();
    sc[tid] = e * inv;            // alpha
    __syncthreads();

    // phase 3: attn_feat[c] = sum_hw featsT[b][hw][c] * alpha[hw]
    const bf16* ft = featsT + (size_t)b * HW_ * C_;
    float acc0 = 0.f, acc1 = 0.f;
    for (int hw = 0; hw < HW_; ++hw) {
        float al = sc[hw];
        acc0 += al * (float)ft[(size_t)hw * C_ + tid];
        acc1 += al * (float)ft[(size_t)hw * C_ + tid + 256];
    }
    x_bf[(size_t)b * INP + tid]       = (bf16)acc0;
    x_bf[(size_t)b * INP + tid + 256] = (bf16)acc1;
    if (tid < 128) {                  // one-hot (embeddings == identity) + zero pad
        int idx = texts[b * T_ + t];
        x_bf[(size_t)b * INP + 512 + tid] = (bf16)((tid == idx) ? 1.f : 0.f);
    }
}

// ---------------- per-step: GI = x@W_ih^T + b_ih ; GH = h@W_hh^T + b_hh ----------------
// 4-way N register blocking: wave = 16(b) x 64(n).  384 waves -> 48 blocks.
__global__ __launch_bounds__(256)
void k_gru(const bf16* __restrict__ x_bf,  const bf16* __restrict__ Wih_bf, const float* __restrict__ b_ih,
           const bf16* __restrict__ h_bf,  const bf16* __restrict__ Whh_bf, const float* __restrict__ b_hh,
           float* __restrict__ GI, float* __restrict__ GH) {
    int tile = blockIdx.x * 8 + (threadIdx.x >> 5);   // 16 b-tiles x 24 n-groups
    int lane = threadIdx.x & 31;
    int b0 = (tile & 15) << 4;
    int n0 = (tile >> 4) << 6;
    int rA = b0 + (lane & 15);
    int rB = n0 + (lane & 15);
    int hi = lane >> 4, n = lane & 15;

    v8f ai[4] = {{}, {}, {}, {}};
#pragma unroll 2
    for (int k = 0; k < INP; k += 32) {
        v16bf a = load_frag(x_bf, INP, rA, k, lane);
#pragma unroll
        for (int j = 0; j < 4; ++j) {
            v16bf bb = load_frag(Wih_bf, INP, rB + 16 * j, k, lane);
            ai[j] = wmma_bf16(a, bb, ai[j]);
        }
    }
#pragma unroll
    for (int j = 0; j < 4; ++j) {
        float bi = b_ih[n0 + 16 * j + n];
#pragma unroll
        for (int r = 0; r < 8; ++r)
            GI[(size_t)(b0 + 8 * hi + r) * G3 + n0 + 16 * j + n] = ai[j][r] + bi;
    }

    v8f ah[4] = {{}, {}, {}, {}};
#pragma unroll 2
    for (int k = 0; k < HID_; k += 32) {
        v16bf a = load_frag(h_bf, HID_, rA, k, lane);
#pragma unroll
        for (int j = 0; j < 4; ++j) {
            v16bf bb = load_frag(Whh_bf, HID_, rB + 16 * j, k, lane);
            ah[j] = wmma_bf16(a, bb, ah[j]);
        }
    }
#pragma unroll
    for (int j = 0; j < 4; ++j) {
        float bi = b_hh[n0 + 16 * j + n];
#pragma unroll
        for (int r = 0; r < 8; ++r)
            GH[(size_t)(b0 + 8 * hi + r) * G3 + n0 + 16 * j + n] = ah[j][r] + bi;
    }
}

// ---------------- per-step: GRU gates + hidden update ----------------
__global__ __launch_bounds__(256)
void k_gates(const float* __restrict__ GI, const float* __restrict__ GH,
             float* __restrict__ h, bf16* __restrict__ h_bf) {
    int i = blockIdx.x * 256 + threadIdx.x;  // B*HID
    int b = i >> 9, j = i & (HID_ - 1);
    size_t base = (size_t)b * G3;
    float r = 1.f / (1.f + __expf(-(GI[base + j] + GH[base + j])));
    float z = 1.f / (1.f + __expf(-(GI[base + HID_ + j] + GH[base + HID_ + j])));
    float nn = fast_tanh(GI[base + 2 * HID_ + j] + r * GH[base + 2 * HID_ + j]);
    float hn = (1.f - z) * nn + z * h[i];
    h[i] = hn;
    h_bf[i] = (bf16)hn;
}

// ---------------- per-step: out[b,t,:] = h_new @ Wg^T + bg ----------------
__global__ __launch_bounds__(256)
void k_gen(const bf16* __restrict__ h_bf, const bf16* __restrict__ Wg_bf,
           const float* __restrict__ bg, float* __restrict__ out, int t) {
    int tile = blockIdx.x * 8 + (threadIdx.x >> 5);   // 112 tiles: 16 b x 7 n
    int lane = threadIdx.x & 31;
    int b0 = (tile & 15) << 4;
    int n0 = (tile >> 4) << 4;
    int rA = b0 + (lane & 15);
    int rB = n0 + (lane & 15);
    v8f acc = {};
#pragma unroll 4
    for (int k = 0; k < HID_; k += 32) {
        v16bf a  = load_frag(h_bf,  HID_, rA, k, lane);
        v16bf bb = load_frag(Wg_bf, HID_, rB, k, lane);
        acc = wmma_bf16(a, bb, acc);
    }
    int hi = lane >> 4, n = lane & 15;
    int nc = n0 + n;
    if (nc < NC_) {
        float bias = bg[nc];
#pragma unroll
        for (int r = 0; r < 8; ++r)
            out[(size_t)(b0 + 8 * hi + r) * (T_ * NC_) + (size_t)t * NC_ + nc] = acc[r] + bias;
    }
}

// ---------------- host orchestration ----------------
extern "C" void kernel_launch(void* const* d_in, const int* in_sizes, int n_in,
                              void* d_out, int out_size, void* d_ws, size_t ws_size,
                              hipStream_t stream) {
    (void)in_sizes; (void)n_in; (void)out_size; (void)ws_size;
    const float* feats = (const float*)d_in[0];
    const int*   texts = (const int*)  d_in[1];
    // d_in[2] = embeddings (identity) -> one-hot generated directly
    const float* Wf   = (const float*)d_in[3];
    const float* bf_  = (const float*)d_in[4];
    const float* Wh   = (const float*)d_in[5];
    const float* bh   = (const float*)d_in[6];
    const float* ws   = (const float*)d_in[7];
    const float* W_ih = (const float*)d_in[8];
    const float* b_ih = (const float*)d_in[9];
    const float* W_hh = (const float*)d_in[10];
    const float* b_hh = (const float*)d_in[11];
    const float* Wg   = (const float*)d_in[12];
    const float* bg   = (const float*)d_in[13];
    float* out = (float*)d_out;

    char* w = (char*)d_ws;
    size_t off = 0;
    auto carve = [&](size_t bytes) -> char* {
        char* p = w + off;
        off += (bytes + 255) & ~size_t(255);
        return p;
    };
    bf16* featsT = (bf16*)carve((size_t)B_ * HW_ * C_ * 2);   // 64 MB (L2 resident)
    bf16* fxT    = (bf16*)carve((size_t)B_ * HW_ * A_ * 2);   // 64 MB (L2 resident)
    bf16* Wf_bf  = (bf16*)carve((size_t)A_ * C_ * 2);
    bf16* Wh_bf  = (bf16*)carve((size_t)A_ * HID_ * 2);
    bf16* Wih_bf = (bf16*)carve((size_t)G3 * INP * 2);
    bf16* Whh_bf = (bf16*)carve((size_t)G3 * HID_ * 2);
    bf16* Wg_bf  = (bf16*)carve((size_t)NCP * HID_ * 2);
    float* h     = (float*)carve((size_t)B_ * HID_ * 4);
    bf16*  h_bf  = (bf16*) carve((size_t)B_ * HID_ * 2);
    float* hq    = (float*)carve((size_t)B_ * A_ * 4);
    bf16*  x_bf  = (bf16*) carve((size_t)B_ * INP * 2);
    float* GI    = (float*)carve((size_t)B_ * G3 * 4);
    float* GH    = (float*)carve((size_t)B_ * G3 * 4);

    // ---- setup ----
    k_convert_pad<<<(A_ * C_   + 255) / 256, 256, 0, stream>>>(Wf,   Wf_bf,  A_,  C_,   A_,  C_);
    k_convert_pad<<<(A_ * HID_ + 255) / 256, 256, 0, stream>>>(Wh,   Wh_bf,  A_,  HID_, A_,  HID_);
    k_convert_pad<<<(G3 * INP  + 255) / 256, 256, 0, stream>>>(W_ih, Wih_bf, G3,  609,  G3,  INP);
    k_convert_pad<<<(G3 * HID_ + 255) / 256, 256, 0, stream>>>(W_hh, Whh_bf, G3,  HID_, G3,  HID_);
    k_convert_pad<<<(NCP * HID_ + 255) / 256, 256, 0, stream>>>(Wg,  Wg_bf,  NC_, HID_, NCP, HID_);
    k_transpose_feats<<<(B_ * HW_ * C_) / 256, 256, 0, stream>>>(feats, featsT);
    k_zero_h<<<(B_ * HID_) / 256, 256, 0, stream>>>(h, h_bf);

    // ---- one-time big GEMM: feat_x (TDM-staged A tile, stored transposed bf16) ----
    k_featx<<<dim3(32, B_), 256, 0, stream>>>(Wf_bf, featsT, bf_, fxT);

    // ---- sequential decode steps ----
    for (int t = 0; t < T_; ++t) {
        k_hq  <<<16,  256, 0, stream>>>(h_bf, Wh_bf, bh, hq);
        k_attn<<<B_,  256, 0, stream>>>(featsT, fxT, hq, ws, texts, t, x_bf);
        k_gru <<<48,  256, 0, stream>>>(x_bf, Wih_bf, b_ih, h_bf, Whh_bf, b_hh, GI, GH);
        k_gates<<<(B_ * HID_) / 256, 256, 0, stream>>>(GI, GH, h, h_bf);
        k_gen <<<14,  256, 0, stream>>>(h_bf, Wg_bf, bg, out, t);
    }
}
